// TrajectoryGNNCollisionPredictor_46823733461642
// MI455X (gfx1250) — matrix operand order repaired and checked by reading, not compiled, not run on hardware
//
#include <hip/hip_runtime.h>
#include <hip/hip_bf16.h>

typedef __bf16 bf16_t;
typedef __attribute__((ext_vector_type(16))) __bf16 v16bf;
typedef __attribute__((ext_vector_type(8)))  __bf16 v8bf;
typedef __attribute__((ext_vector_type(8)))  float  v8f;

#define NNODE 1536
#define TT 10
#define DD 64
#define NPAIR 1178880   // NNODE*(NNODE-1)/2
#define GTILES 2        // M-tiles (16 pairs each) per wave in edge_mlp

// gfx1250 async memory->LDS path (guarded so absence degrades, not fails)
#if defined(__has_builtin)
#  if __has_builtin(__builtin_amdgcn_global_load_async_to_lds_b128) && \
      __has_builtin(__builtin_amdgcn_s_wait_asynccnt)
#    define USE_ASYNC_LDS 1
#  endif
#endif
#ifndef USE_ASYNC_LDS
#  define USE_ASYNC_LDS 0
#endif

#if USE_ASYNC_LDS
// Parameter type per hipcc diagnostic: int __vector_size__(16) __device__ *
typedef int v4i_async __attribute__((vector_size(16)));
#define ASYNC_GPTR(p) ((__attribute__((address_space(1))) v4i_async*)(p))
#define ASYNC_LPTR(p) ((__attribute__((address_space(3))) v4i_async*)(p))
#endif

__device__ __forceinline__ v8f wmma_bf16(v16bf a, v16bf b, v8f c) {
  // D = A(16x32 bf16) * B(32x16 bf16) + C(16x16 f32)
  return __builtin_amdgcn_wmma_f32_16x16x32_bf16(false, a, false, b, (short)0, c, false, false);
}

union AFrag { v16bf v; v8bf h[2]; bf16_t e[16]; };

// ---------------------------------------------------------------------------
// Kernel 1: per-node transformer encoder (post-norm, relu), mean over T.
// One block (64 threads) per node. Writes node features as bf16.
// ---------------------------------------------------------------------------
__global__ void __launch_bounds__(64) traj_encoder(
    const float* __restrict__ traj,
    const float* __restrict__ w_in, const float* __restrict__ b_in,
    const float* __restrict__ w_out, const float* __restrict__ b_out,
    const float* __restrict__ ln1g, const float* __restrict__ ln1b,
    const float* __restrict__ fw1, const float* __restrict__ fb1,
    const float* __restrict__ fw2, const float* __restrict__ fb2,
    const float* __restrict__ ln2g, const float* __restrict__ ln2b,
    bf16_t* __restrict__ node_out)
{
  const int n = blockIdx.x;
  const int tid = threadIdx.x;
  __shared__ float x[TT][DD];
  __shared__ float qkv[TT][3 * DD];
  __shared__ float att[4][TT][TT];
  __shared__ float buf[TT][DD];
  __shared__ float y[TT][DD];

  for (int i = tid; i < TT * DD; i += 64) x[i / DD][i % DD] = traj[n * TT * DD + i];
  __syncthreads();

  // QKV projection: qkv = x @ w_in.T + b_in   (w_in: (192,64))
  for (int o = tid; o < 3 * DD; o += 64) {
    const float* wr = w_in + o * DD;
    for (int t = 0; t < TT; ++t) {
      float acc = b_in[o];
      #pragma unroll 8
      for (int c = 0; c < DD; ++c) acc += x[t][c] * wr[c];
      qkv[t][o] = acc;
    }
  }
  __syncthreads();

  // attention scores + softmax: one thread per (head, t)
  if (tid < 4 * TT) {
    int h = tid / TT, t = tid % TT;
    float sc[TT];
    float mx = -1e30f;
    for (int s = 0; s < TT; ++s) {
      float a = 0.f;
      #pragma unroll
      for (int d = 0; d < 16; ++d) a += qkv[t][h * 16 + d] * qkv[s][DD + h * 16 + d];
      a *= 0.25f;  // 1/sqrt(16)
      sc[s] = a; mx = fmaxf(mx, a);
    }
    float sum = 0.f;
    for (int s = 0; s < TT; ++s) { sc[s] = __expf(sc[s] - mx); sum += sc[s]; }
    float inv = 1.f / sum;
    for (int s = 0; s < TT; ++s) att[h][t][s] = sc[s] * inv;
  }
  __syncthreads();

  // ctx = attn @ v   (one thread per channel)
  {
    int c = tid, h = c >> 4;
    for (int t = 0; t < TT; ++t) {
      float a = 0.f;
      for (int s = 0; s < TT; ++s) a += att[h][t][s] * qkv[s][2 * DD + c];
      buf[t][c] = a;
    }
  }
  __syncthreads();

  // output projection + residual
  {
    int c = tid;
    const float* wr = w_out + c * DD;
    for (int t = 0; t < TT; ++t) {
      float a = b_out[c];
      #pragma unroll 8
      for (int k = 0; k < DD; ++k) a += buf[t][k] * wr[k];
      y[t][c] = x[t][c] + a;
    }
  }
  __syncthreads();

  // LayerNorm 1 -> x
  if (tid < TT) {
    float m = 0.f;
    for (int c = 0; c < DD; ++c) m += y[tid][c];
    m *= (1.f / DD);
    float v = 0.f;
    for (int c = 0; c < DD; ++c) { float d = y[tid][c] - m; v += d * d; }
    v *= (1.f / DD);
    float inv = rsqrtf(v + 1e-5f);
    for (int c = 0; c < DD; ++c) x[tid][c] = (y[tid][c] - m) * inv * ln1g[c] + ln1b[c];
  }
  __syncthreads();

  // FFN layer 1 (relu) -> buf
  {
    int c = tid;
    const float* wr = fw1 + c * DD;
    for (int t = 0; t < TT; ++t) {
      float a = fb1[c];
      #pragma unroll 8
      for (int k = 0; k < DD; ++k) a += x[t][k] * wr[k];
      buf[t][c] = fmaxf(a, 0.f);
    }
  }
  __syncthreads();

  // FFN layer 2 + residual -> y
  {
    int c = tid;
    const float* wr = fw2 + c * DD;
    for (int t = 0; t < TT; ++t) {
      float a = fb2[c];
      #pragma unroll 8
      for (int k = 0; k < DD; ++k) a += buf[t][k] * wr[k];
      y[t][c] = x[t][c] + a;
    }
  }
  __syncthreads();

  // LayerNorm 2 -> x
  if (tid < TT) {
    float m = 0.f;
    for (int c = 0; c < DD; ++c) m += y[tid][c];
    m *= (1.f / DD);
    float v = 0.f;
    for (int c = 0; c < DD; ++c) { float d = y[tid][c] - m; v += d * d; }
    v *= (1.f / DD);
    float inv = rsqrtf(v + 1e-5f);
    for (int c = 0; c < DD; ++c) x[tid][c] = (y[tid][c] - m) * inv * ln2g[c] + ln2b[c];
  }
  __syncthreads();

  // mean over T -> node feature (bf16)
  {
    int c = tid;
    float s = 0.f;
    for (int t = 0; t < TT; ++t) s += x[t][c];
    node_out[n * DD + c] = (bf16_t)(s * 0.1f);
  }
}

// ---------------------------------------------------------------------------
// Kernel 2: pack edge-MLP + GCN weight matrices into WMMA B-fragment order.
// Fragment element layout: ((frag)*32 + lane)*16 + e, where lane holds
// col = n*16 + (lane&15), k = 32*s + 8*(lane>>4) + e + (e&8).
// ---------------------------------------------------------------------------
__global__ void __launch_bounds__(256) prep_weights(
    const float* __restrict__ e_w1,   // (128,138)
    const float* __restrict__ e_w2,   // (64,128)
    const float* __restrict__ gcn_w,  // (3,64,64)
    bf16_t* __restrict__ w1f,         // 5*8 frags (20480 elems)
    bf16_t* __restrict__ w2f,         // 4*4 frags (8192 elems)
    bf16_t* __restrict__ gwf)         // 3*(2*4) frags (12288 elems)
{
  int idx = blockIdx.x * 256 + threadIdx.x;
  if (idx < 20480) {  // w1: K padded 138 -> 160 (s<5), 8 col-tiles
    int e = idx & 15, lane = (idx >> 4) & 31, fn = idx >> 9;
    int n = fn & 7, s = fn >> 3;
    int col = n * 16 + (lane & 15);
    int k = 32 * s + 8 * (lane >> 4) + e + (e & 8);
    float v = (k < 138) ? e_w1[col * 138 + k] : 0.f;
    w1f[idx] = (bf16_t)v;
  }
  int idx2 = idx - 20480;
  if (idx2 >= 0 && idx2 < 8192) {  // w2: K=128 (s<4), 4 col-tiles
    int e = idx2 & 15, lane = (idx2 >> 4) & 31, fn = idx2 >> 9;
    int n = fn & 3, s = fn >> 2;
    int col = n * 16 + (lane & 15);
    int k = 32 * s + 8 * (lane >> 4) + e + (e & 8);
    w2f[idx2] = (bf16_t)e_w2[col * 128 + k];
  }
  int idx3 = idx - 28672;
  if (idx3 >= 0 && idx3 < 12288) {  // gcn: 3 layers, K=64 (s<2), 4 col-tiles
    int e = idx3 & 15, lane = (idx3 >> 4) & 31, fn = idx3 >> 9;
    int l = fn >> 3, rem = fn & 7, n = rem & 3, s = rem >> 2;
    int col = n * 16 + (lane & 15);
    int k = 32 * s + 8 * (lane >> 4) + e + (e & 8);
    gwf[idx3] = (bf16_t)gcn_w[(l * 64 + col) * 64 + k];
  }
}

// ---------------------------------------------------------------------------
// Kernel 3: transpose node features (row-major bf16 -> column-major bf16)
// so gcn_agg B-fragment loads are 32B contiguous.
// ---------------------------------------------------------------------------
__global__ void __launch_bounds__(256) transpose_node(
    const bf16_t* __restrict__ node, bf16_t* __restrict__ node_cm)
{
  int idx = blockIdx.x * 256 + threadIdx.x;
  if (idx < NNODE * DD) {
    int r = idx >> 6, c = idx & 63;
    node_cm[(size_t)c * NNODE + r] = node[idx];
  }
}

// ---------------------------------------------------------------------------
// Kernel 4: agg = adj @ node  via bf16 WMMA, K-loop over 1536.
// One wave computes a 16x64 output tile. 12 blocks * 8 waves = 96 tiles.
// ---------------------------------------------------------------------------
__global__ void __launch_bounds__(256) gcn_agg(
    const float* __restrict__ adj,        // (1536,1536) f32
    const bf16_t* __restrict__ node_cm,   // (64,1536) bf16 column-major
    bf16_t* __restrict__ agg)             // (1536,64) bf16
{
  const int lane = threadIdx.x & 31;
  const int wave = threadIdx.x >> 5;
  const int m0 = (blockIdx.x * 8 + wave) * 16;
  const int row = lane & 15;
  const int half = lane >> 4;

  v8f zero = {};
  v8f acc[4];
  #pragma unroll
  for (int n = 0; n < 4; ++n) acc[n] = zero;

  const float* arow = adj + (size_t)(m0 + row) * NNODE;
  for (int s = 0; s < 48; ++s) {
    AFrag a;
    int k0 = 32 * s + 16 * half;
    #pragma unroll
    for (int q = 0; q < 8; ++q) a.e[q] = (bf16_t)arow[k0 + q];
    #pragma unroll
    for (int q = 0; q < 8; ++q) a.e[8 + q] = (bf16_t)arow[k0 + 8 + q];
    #pragma unroll
    for (int n = 0; n < 4; ++n) {
      int col = n * 16 + row;
      v16bf b = *(const v16bf*)(node_cm + (size_t)col * NNODE + 32 * s + 16 * half);
      acc[n] = wmma_bf16(a.v, b, acc[n]);
    }
  }
  #pragma unroll
  for (int n = 0; n < 4; ++n) {
    int col = n * 16 + row;
    #pragma unroll
    for (int r = 0; r < 8; ++r)
      agg[(size_t)(m0 + r + 8 * half) * DD + col] = (bf16_t)acc[n][r];
  }
}

// ---------------------------------------------------------------------------
// Kernel 5: node_out = relu(agg @ W^T + b)  via bf16 WMMA (K=64).
// ---------------------------------------------------------------------------
__global__ void __launch_bounds__(256) gcn_lin(
    const bf16_t* __restrict__ agg,   // (1536,64) bf16
    const bf16_t* __restrict__ wf,    // packed frags for this layer (2*4)
    const float* __restrict__ bias,   // 64
    bf16_t* __restrict__ node_out)    // (1536,64) bf16
{
  const int lane = threadIdx.x & 31;
  const int wave = threadIdx.x >> 5;
  const int m0 = (blockIdx.x * 8 + wave) * 16;
  const int row = lane & 15;
  const int half = lane >> 4;

  v8f zero = {};
  v8f acc[4];
  #pragma unroll
  for (int n = 0; n < 4; ++n) acc[n] = zero;

  const bf16_t* ar = agg + (size_t)(m0 + row) * DD;
  #pragma unroll
  for (int s = 0; s < 2; ++s) {
    AFrag a;
    a.h[0] = *(const v8bf*)(ar + 32 * s + 8 * half);
    a.h[1] = *(const v8bf*)(ar + 32 * s + 16 + 8 * half);
    #pragma unroll
    for (int n = 0; n < 4; ++n) {
      v16bf b = *(const v16bf*)(wf + ((s * 4 + n) * 32 + lane) * 16);
      acc[n] = wmma_bf16(a.v, b, acc[n]);
    }
  }
  #pragma unroll
  for (int n = 0; n < 4; ++n) {
    int col = n * 16 + row;
    float bb = bias[col];
    #pragma unroll
    for (int r = 0; r < 8; ++r)
      node_out[(size_t)(m0 + r + 8 * half) * DD + col] = (bf16_t)fmaxf(acc[n][r] + bb, 0.f);
  }
}

// ---------------------------------------------------------------------------
// Kernel 6: all-pairs edge predictor.
// One wave = 2 M-tiles = 32 pairs; one block = 8 waves = 256 pairs.
// W1/W2 fragments staged once per block into LDS (async-to-LDS when the
// toolchain exposes it); each B-fragment LDS read feeds GTILES wmmas.
// hist output staged in LDS and written fully coalesced.
// ---------------------------------------------------------------------------
__global__ void __launch_bounds__(256) edge_mlp(
    const float* __restrict__ traj,     // (1536,10,64)
    const bf16_t* __restrict__ node,    // (1536,64) final GCN features
    const bf16_t* __restrict__ w1f, const float* __restrict__ b1,
    const bf16_t* __restrict__ w2f, const float* __restrict__ b2,
    const float* __restrict__ w3, const float* __restrict__ b3,
    float* __restrict__ out)            // [prob P][min_dist P][hist P*10]
{
  const int tid = threadIdx.x;
  const int lane = tid & 31;
  const int wave = tid >> 5;
  const int row = lane & 15;
  const int half = lane >> 4;
  const int p0 = blockIdx.x * (8 * GTILES * 16) + wave * (GTILES * 16);

  __shared__ bf16_t w1s[20480];                             // 40 KB
  __shared__ bf16_t w2s[8192];                              // 16 KB
  __shared__ __align__(32) bf16_t feat[8][GTILES][16][160]; // 80 KB
  __shared__ float histb[8][GTILES][16][TT];                // 10 KB
  __shared__ int sii[8][GTILES][16];
  __shared__ int sjj[8][GTILES][16];

  // --- stage weight fragments into LDS (whole block cooperates)
#if USE_ASYNC_LDS
  {
    char* g1 = (char*)w1f;
    char* l1 = (char*)w1s;
    for (int o = tid; o < 40960 / 16; o += 256)
      __builtin_amdgcn_global_load_async_to_lds_b128(
          ASYNC_GPTR(g1 + o * 16), ASYNC_LPTR(l1 + o * 16), 0, 0);
    char* g2 = (char*)w2f;
    char* l2 = (char*)w2s;
    for (int o = tid; o < 16384 / 16; o += 256)
      __builtin_amdgcn_global_load_async_to_lds_b128(
          ASYNC_GPTR(g2 + o * 16), ASYNC_LPTR(l2 + o * 16), 0, 0);
    __builtin_amdgcn_s_wait_asynccnt(0);
  }
#else
  for (int o = tid; o < 20480 / 8; o += 256)
    *(v8bf*)(w1s + o * 8) = *(const v8bf*)(w1f + o * 8);
  for (int o = tid; o < 8192 / 8; o += 256)
    *(v8bf*)(w2s + o * 8) = *(const v8bf*)(w2f + o * 8);
#endif

  // --- phase 1: each lane owns pair (g = half, row): invert triangular index,
  //     compute hist + min_dist, write min_dist + feature tail.
  const int gown = half;
  int p = p0 + gown * 16 + row;
  bool valid = p < NPAIR;
  int pc = valid ? p : (NPAIR - 1);
  const float t2 = 2.f * NNODE - 1.f;                 // 3071
  int i = (int)floorf((t2 - sqrtf(t2 * t2 - 8.f * (float)pc)) * 0.5f);
  if (i < 0) i = 0;
  if (i > NNODE - 2) i = NNODE - 2;
  #define ROWSTART(a) ((a) * (2 * NNODE - 1 - (a)) / 2)
  while (ROWSTART(i + 1) <= pc) ++i;
  while (ROWSTART(i) > pc) --i;
  int j = i + 1 + (pc - ROWSTART(i));
  #undef ROWSTART
  sii[wave][gown][row] = i;
  sjj[wave][gown][row] = j;
  // prefetch the node rows needed in phase 2 (global_prefetch_b8)
  __builtin_prefetch(node + (size_t)i * DD, 0, 0);
  __builtin_prefetch(node + (size_t)j * DD, 0, 0);
  {
    float mind = 1e30f;
    const float* ti = traj + (size_t)i * (TT * DD);
    const float* tj = traj + (size_t)j * (TT * DD);
    #pragma unroll
    for (int t = 0; t < TT; ++t) {
      float2 pi = *(const float2*)(ti + t * DD);
      float2 pj = *(const float2*)(tj + t * DD);
      float dx = pi.x - pj.x;
      float dy = pi.y - pj.y;
      float d = sqrtf(dx * dx + dy * dy);
      mind = fminf(mind, d);
      float hv = 1.f / (1.f + __expf(-10.f * (d - 5.f)));
      feat[wave][gown][row][128 + t] = (bf16_t)hv;
      histb[wave][gown][row][t] = hv;
    }
    #pragma unroll
    for (int t = 138; t < 160; ++t) feat[wave][gown][row][t] = (bf16_t)0.f;
    if (valid) out[(size_t)NPAIR + p] = mind;
  }
  __syncthreads();

  // --- coalesced hist writes: 320 contiguous floats per wave
  {
    const float* hb = &histb[wave][0][0][0];
    size_t base = (size_t)2 * NPAIR + (size_t)p0 * TT;
    for (int o = lane; o < GTILES * 16 * TT; o += 32) {
      int pp = p0 + o / TT;
      if (pp < NPAIR) out[base + o] = hb[o];
    }
  }

  // --- phase 2: node feature copies (half 0 -> node[i], half 1 -> node[j])
  #pragma unroll
  for (int g = 0; g < GTILES; ++g) {
    int ii = half ? sjj[wave][g][row] : sii[wave][g][row];
    const bf16_t* src = node + (size_t)ii * DD;
    bf16_t* dst = &feat[wave][g][row][half * DD];
    #pragma unroll
    for (int c = 0; c < DD; c += 8) *(v8bf*)(dst + c) = *(const v8bf*)(src + c);
  }
  __syncthreads();

  // --- GEMM1: (16x160) @ (160x128) per tile; B frag from LDS reused GTILES x
  v8f zero = {};
  v8f acc[GTILES][8];
  #pragma unroll
  for (int g = 0; g < GTILES; ++g)
    #pragma unroll
    for (int n = 0; n < 8; ++n) acc[g][n] = zero;
  #pragma unroll
  for (int s = 0; s < 5; ++s) {
    AFrag a[GTILES];
    #pragma unroll
    for (int g = 0; g < GTILES; ++g) {
      const bf16_t* frow = &feat[wave][g][row][0];
      a[g].h[0] = *(const v8bf*)(frow + 32 * s + 8 * half);
      a[g].h[1] = *(const v8bf*)(frow + 32 * s + 16 + 8 * half);
    }
    #pragma unroll
    for (int n = 0; n < 8; ++n) {
      v16bf b = *(const v16bf*)(w1s + ((s * 8 + n) * 32 + lane) * 16);
      #pragma unroll
      for (int g = 0; g < GTILES; ++g) acc[g][n] = wmma_bf16(a[g].v, b, acc[g][n]);
    }
  }
  __syncthreads();

  // bias + relu -> h1 (bf16) in the same LDS tiles (cols 0..127)
  #pragma unroll
  for (int n = 0; n < 8; ++n) {
    int col = n * 16 + row;
    float bb = b1[col];
    #pragma unroll
    for (int g = 0; g < GTILES; ++g)
      #pragma unroll
      for (int r = 0; r < 8; ++r)
        feat[wave][g][r + 8 * half][col] = (bf16_t)fmaxf(acc[g][n][r] + bb, 0.f);
  }
  __syncthreads();

  // --- GEMM2: (16x128) @ (128x64) per tile
  v8f acc2[GTILES][4];
  #pragma unroll
  for (int g = 0; g < GTILES; ++g)
    #pragma unroll
    for (int n = 0; n < 4; ++n) acc2[g][n] = zero;
  #pragma unroll
  for (int s = 0; s < 4; ++s) {
    AFrag a[GTILES];
    #pragma unroll
    for (int g = 0; g < GTILES; ++g) {
      const bf16_t* frow = &feat[wave][g][row][0];
      a[g].h[0] = *(const v8bf*)(frow + 32 * s + 8 * half);
      a[g].h[1] = *(const v8bf*)(frow + 32 * s + 16 + 8 * half);
    }
    #pragma unroll
    for (int n = 0; n < 4; ++n) {
      v16bf b = *(const v16bf*)(w2s + ((s * 4 + n) * 32 + lane) * 16);
      #pragma unroll
      for (int g = 0; g < GTILES; ++g) acc2[g][n] = wmma_bf16(a[g].v, b, acc2[g][n]);
    }
  }
  __syncthreads();

  // bias + relu -> h2 (f32, 16x64 per tile) reusing the LDS tiles
  #pragma unroll
  for (int n = 0; n < 4; ++n) {
    int col = n * 16 + row;
    float bb = b2[col];
    #pragma unroll
    for (int g = 0; g < GTILES; ++g) {
      float* h2 = (float*)&feat[wave][g][0][0];
      #pragma unroll
      for (int r = 0; r < 8; ++r)
        h2[(r + 8 * half) * DD + col] = fmaxf(acc2[g][n][r] + bb, 0.f);
    }
  }
  __syncthreads();

  // --- final 64-dot + sigmoid: lane handles its own pair (g = half, row)
  {
    const float* h2 = (const float*)&feat[wave][gown][0][0];
    float s = b3[0];
    #pragma unroll 8
    for (int c = 0; c < DD; ++c) s += h2[row * DD + c] * w3[c];
    float pr = 1.f / (1.f + __expf(-s));
    if (valid) out[p] = pr;
  }
}

// ---------------------------------------------------------------------------
extern "C" void kernel_launch(void* const* d_in, const int* in_sizes, int n_in,
                              void* d_out, int out_size, void* d_ws, size_t ws_size,
                              hipStream_t stream) {
  (void)in_sizes; (void)n_in; (void)out_size; (void)ws_size;
  const float* traj  = (const float*)d_in[0];
  const float* adj   = (const float*)d_in[1];
  const float* aiw   = (const float*)d_in[2];
  const float* aib   = (const float*)d_in[3];
  const float* aow   = (const float*)d_in[4];
  const float* aob   = (const float*)d_in[5];
  const float* ln1g  = (const float*)d_in[6];
  const float* ln1b  = (const float*)d_in[7];
  const float* ffw1  = (const float*)d_in[8];
  const float* ffb1  = (const float*)d_in[9];
  const float* ffw2  = (const float*)d_in[10];
  const float* ffb2  = (const float*)d_in[11];
  const float* ln2g  = (const float*)d_in[12];
  const float* ln2b  = (const float*)d_in[13];
  const float* gcnw  = (const float*)d_in[14];
  const float* gcnb  = (const float*)d_in[15];
  const float* ew1   = (const float*)d_in[16];
  const float* eb1   = (const float*)d_in[17];
  const float* ew2   = (const float*)d_in[18];
  const float* eb2   = (const float*)d_in[19];
  const float* ew3   = (const float*)d_in[20];
  const float* eb3   = (const float*)d_in[21];
  float* out = (float*)d_out;

  char* ws = (char*)d_ws;
  bf16_t* nodeA  = (bf16_t*)(ws + 0);        // 196608 B
  bf16_t* nodeB  = (bf16_t*)(ws + 196608);   // 196608 B
  bf16_t* nodeCM = (bf16_t*)(ws + 393216);   // 196608 B
  bf16_t* agg    = (bf16_t*)(ws + 589824);   // 196608 B
  bf16_t* w1f    = (bf16_t*)(ws + 786432);   // 40960 B
  bf16_t* w2f    = (bf16_t*)(ws + 827392);   // 16384 B
  bf16_t* gwf    = (bf16_t*)(ws + 843776);   // 24576 B

  traj_encoder<<<NNODE, 64, 0, stream>>>(traj, aiw, aib, aow, aob, ln1g, ln1b,
                                         ffw1, ffb1, ffw2, ffb2, ln2g, ln2b, nodeA);
  prep_weights<<<160, 256, 0, stream>>>(ew1, ew2, gcnw, w1f, w2f, gwf);

  bf16_t* cur = nodeA; bf16_t* nxt = nodeB;
  for (int l = 0; l < 3; ++l) {
    transpose_node<<<384, 256, 0, stream>>>(cur, nodeCM);
    gcn_agg<<<12, 256, 0, stream>>>(adj, nodeCM, agg);
    gcn_lin<<<12, 256, 0, stream>>>(agg, gwf + l * 4096, gcnb + l * 64, nxt);
    bf16_t* t = cur; cur = nxt; nxt = t;
  }

  // P = 4605 * 256 pairs exactly
  edge_mlp<<<NPAIR / 256, 256, 0, stream>>>(traj, cur, w1f, eb1, w2f, eb2,
                                            ew3, eb3, out);
}
